// TransformerLayer_36799279792716
// MI455X (gfx1250) — matrix operand order repaired and checked by reading, not compiled, not run on hardware
//
#include <hip/hip_runtime.h>
#include <hip/hip_bf16.h>

// ---------------------------------------------------------------------------
// Problem constants (from reference): B,S,D,H,DH,DI,K = 2,2048,1024,16,64,4096,3
// ---------------------------------------------------------------------------
constexpr int cB  = 2;
constexpr int cS  = 2048;
constexpr int cD  = 1024;
constexpr int cH  = 16;
constexpr int cDH = 64;
constexpr int cDI = 4096;
constexpr int cHD = cH * cDH;       // 1024
constexpr int cBS = cB * cS;        // 4096

typedef __attribute__((ext_vector_type(16))) __bf16 v16bf;
typedef __attribute__((ext_vector_type(8)))  float  v8f;
typedef int v4i __attribute__((vector_size(16)));   // matches builtin's V4i

#define AS1 __attribute__((address_space(1)))
#define AS3 __attribute__((address_space(3)))

#if __has_builtin(__builtin_amdgcn_global_load_async_to_lds_b128)
#define GEMM_ASYNC 1
#else
#define GEMM_ASYNC 0
#endif

union BF16x16 { v16bf v; uint4 q[2]; };

__device__ __forceinline__ __bf16 f2bf(float f) {
  unsigned u = __builtin_bit_cast(unsigned, f);
  u += 0x7FFFu + ((u >> 16) & 1u);          // round-to-nearest-even
  unsigned short h = (unsigned short)(u >> 16);
  return __builtin_bit_cast(__bf16, h);
}
__device__ __forceinline__ __bf16 bf_zero() {
  return __builtin_bit_cast(__bf16, (unsigned short)0);
}

__device__ __forceinline__ v8f wmma_bf16(v16bf a, v16bf b, v8f c) {
  // D = A(16x32 bf16) x B(32x16 bf16) + C(16x16 f32)
  return __builtin_amdgcn_wmma_f32_16x16x32_bf16(false, a, false, b,
                                                 (short)0, c, false, false);
}

__device__ __forceinline__ void wait_asynccnt0() {
#if __has_builtin(__builtin_amdgcn_s_wait_asynccnt)
  __builtin_amdgcn_s_wait_asynccnt(0);
#else
  asm volatile("s_wait_asynccnt 0" ::: "memory");
#endif
}

#if GEMM_ASYNC
__device__ __forceinline__ void async_copy16(const __bf16* gsrc, __bf16* lds) {
  __builtin_amdgcn_global_load_async_to_lds_b128(
      (AS1 v4i*)gsrc, (AS3 v4i*)lds, 0, 0);
}
#endif

// A-fragment (16x32, row m = lane&15). halves 0..7 : K = k0 + (lane<16?0:8)+i
//                                      halves 8..15: K = k0+16+(lane<16?0:8)+i
__device__ __forceinline__ v16bf load_frag_a(const __bf16* __restrict__ A,
                                             long lda, int m, int k0, int lane) {
  const __bf16* row = A + (long)m * lda;
  int kh = (lane >> 4) << 3;
  BF16x16 u;
  u.q[0] = *(const uint4*)(row + k0 + kh);
  u.q[1] = *(const uint4*)(row + k0 + 16 + kh);
  return u.v;
}
// B-fragment (32x16). Bt is [N,K] row-major; column n = lane&15,
// halves 0..7 : K = k0 + (lane<16?0:8)+i ; halves 8..15: K = k0+16+(lane<16?0:8)+i
__device__ __forceinline__ v16bf load_frag_b(const __bf16* __restrict__ Bt,
                                             long ldb, int n, int k0, int lane) {
  const __bf16* row = Bt + (long)n * ldb;
  int kh = (lane >> 4) << 3;
  BF16x16 u;
  u.q[0] = *(const uint4*)(row + k0 + kh);
  u.q[1] = *(const uint4*)(row + k0 + 16 + kh);
  return u.v;
}

// ---------------------------------------------------------------------------
// Generic bf16 WMMA GEMM:  C[M,N] = A[M,K] * Bt[N,K]^T (+bias)(relu)
// block = 256 thr = 8 waves, 2(M) x 4(N); block tile 64x256; wave tile 32x64
// (2 A-frags x 4 B-frags -> 8 WMMA / K-step, 1.5 vmem loads per WMMA).
// A tile (64x32 = 4KB/K-step) is double-buffered in LDS via
// GLOBAL_LOAD_ASYNC_TO_LDS_B128 when available (ASYNCcnt + barrier protocol).
// Requirements (all call sites satisfy): M%64==0, N%256==0, K%32==0.
// ---------------------------------------------------------------------------
__global__ __launch_bounds__(256)
void gemm_bf16_wmma(const __bf16* __restrict__ A, long lda, long strideAb,
                    const __bf16* __restrict__ Bt, long ldb,
                    const float* __restrict__ bias,
                    float* __restrict__ Cf, long ldc, long strideCfb,
                    __bf16* __restrict__ Cb, long ldcb, long strideCbb,
                    int M, int N, int Kd, int relu)
{
  int lane = threadIdx.x & 31;
  int w    = threadIdx.x >> 5;
  int wm = w & 1;                    // 2 waves in M
  int wn = w >> 1;                   // 4 waves in N
  int mblk = blockIdx.y * 64;
  int m0   = mblk + wm * 32;
  int n0   = blockIdx.x * 256 + wn * 64;
  long zb  = blockIdx.z;
  const __bf16* Ab = A + zb * strideAb;

  int nc[4];
#pragma unroll
  for (int i = 0; i < 4; ++i) nc[i] = n0 + i * 16 + (lane & 15);

  v8f acc[2][4] = {};

#if GEMM_ASYNC
  __shared__ __bf16 atile[2][64][32];
  int tid  = (int)threadIdx.x;
  int crow = tid >> 2;               // 0..63 : row of A tile this lane copies
  int ccol = (tid & 3) << 3;         // 0,8,16,24 : 8-elem (16B) chunk
  const __bf16* gsrc = Ab + (long)(mblk + crow) * lda + ccol;

  // prologue: async-copy K-tile 0 into buffer 0
  async_copy16(gsrc, &atile[0][crow][ccol]);

  int arow0 = wm * 32 + (lane & 15);
  int nk = Kd >> 5;
  for (int kb = 0; kb < nk; ++kb) {
    int k0  = kb << 5;
    int cur = kb & 1;
    wait_asynccnt0();                // our slice of buffer `cur` has landed
    __syncthreads();                 // everyone's slice landed; prev buf free
    if (kb + 1 < nk)
      async_copy16(gsrc + k0 + 32, &atile[cur ^ 1][crow][ccol]);

    int kh = (lane >> 4) << 3;
    BF16x16 a0, a1;
    a0.q[0] = *(const uint4*)&atile[cur][arow0][kh];
    a0.q[1] = *(const uint4*)&atile[cur][arow0][16 + kh];
    a1.q[0] = *(const uint4*)&atile[cur][arow0 + 16][kh];
    a1.q[1] = *(const uint4*)&atile[cur][arow0 + 16][16 + kh];

    v16bf bfr[4];
#pragma unroll
    for (int i = 0; i < 4; ++i) bfr[i] = load_frag_b(Bt, ldb, nc[i], k0, lane);
#pragma unroll
    for (int i = 0; i < 4; ++i) {
      acc[0][i] = wmma_bf16(a0.v, bfr[i], acc[0][i]);
      acc[1][i] = wmma_bf16(a1.v, bfr[i], acc[1][i]);
    }
  }
#else
  int mr0 = m0 + (lane & 15);
  int mr1 = mr0 + 16;
  for (int k0 = 0; k0 < Kd; k0 += 32) {
    v16bf a0 = load_frag_a(Ab, lda, mr0, k0, lane);
    v16bf a1 = load_frag_a(Ab, lda, mr1, k0, lane);
    v16bf bfr[4];
#pragma unroll
    for (int i = 0; i < 4; ++i) bfr[i] = load_frag_b(Bt, ldb, nc[i], k0, lane);
#pragma unroll
    for (int i = 0; i < 4; ++i) {
      acc[0][i] = wmma_bf16(a0, bfr[i], acc[0][i]);
      acc[1][i] = wmma_bf16(a1, bfr[i], acc[1][i]);
    }
  }
#endif
  (void)M; (void)N;

  int rbase0 = m0 + ((lane >> 4) << 3);
#pragma unroll
  for (int g = 0; g < 2; ++g) {
    int rbase = rbase0 + g * 16;
#pragma unroll
    for (int i = 0; i < 4; ++i) {
      float bi = bias ? bias[nc[i]] : 0.f;
#pragma unroll
      for (int r = 0; r < 8; ++r) {
        float v = acc[g][i][r] + bi;
        if (relu) v = fmaxf(v, 0.f);
        long row = rbase + r;
        if (Cf) Cf[zb * strideCfb + row * ldc + nc[i]] = v;
        if (Cb) Cb[zb * strideCbb + row * ldcb + nc[i]] = f2bf(v);
      }
    }
  }
}

// ---------------------------------------------------------------------------
// Flash attention: one wave owns 16 query rows x full DH=64; streams 32-wide
// key blocks with online softmax. Output written with the torch-view quirk.
// ---------------------------------------------------------------------------
__global__ __launch_bounds__(256)
void attn_flash(const __bf16* __restrict__ qb, const __bf16* __restrict__ kb,
                const __bf16* __restrict__ vt, const unsigned char* __restrict__ mk,
                __bf16* __restrict__ av, float scale)
{
  int lane = threadIdx.x & 31;
  int w    = threadIdx.x >> 5;
  int h = blockIdx.y, b = blockIdx.z;
  int q0 = (blockIdx.x * 8 + w) * 16;

  __shared__ __bf16 plds[8][16][32];

  // Q fragments (two 16x32 A-frags covering DH=64)
  v16bf qa[2];
  {
    const __bf16* row = qb + ((long)(b * cS + q0 + (lane & 15))) * cHD + h * cDH;
    int kh = (lane >> 4) << 3;
    BF16x16 u0, u1;
    u0.q[0] = *(const uint4*)(row + kh);
    u0.q[1] = *(const uint4*)(row + 16 + kh);
    u1.q[0] = *(const uint4*)(row + 32 + kh);
    u1.q[1] = *(const uint4*)(row + 48 + kh);
    qa[0] = u0.v; qa[1] = u1.v;
  }

  float mrow[8], lrow[8];
  v8f o[4] = {};
#pragma unroll
  for (int r = 0; r < 8; ++r) { mrow[r] = -1e30f; lrow[r] = 0.f; }

  const __bf16* kbase = kb + ((long)b * cS) * cHD + h * cDH;
  const __bf16* vtb   = vt + ((long)(b * cH + h)) * cDH * cS;
  const unsigned char* mrow8 = mk + (long)b * cS;

  for (int j = 0; j < cS; j += 32) {
    // ---- scores: two 16x16 C-frags, K-dim = DH = 64 (2 WMMA each) ----
    v8f sfr[2];
#pragma unroll
    for (int t = 0; t < 2; ++t) {
      int key = j + t * 16 + (lane & 15);
      const __bf16* krow = kbase + (long)key * cHD;
      int kh = (lane >> 4) << 3;
      BF16x16 u0, u1;
      u0.q[0] = *(const uint4*)(krow + kh);
      u0.q[1] = *(const uint4*)(krow + 16 + kh);
      u1.q[0] = *(const uint4*)(krow + 32 + kh);
      u1.q[1] = *(const uint4*)(krow + 48 + kh);
      v8f acc = {};
      acc = wmma_bf16(qa[0], u0.v, acc);
      acc = wmma_bf16(qa[1], u1.v, acc);
      sfr[t] = acc;
    }

    float pen0 = mrow8[j + (lane & 15)]      ? 0.f : -100000.0f;
    float pen1 = mrow8[j + 16 + (lane & 15)] ? 0.f : -100000.0f;

    float x0[8], x1[8], corr[8];
#pragma unroll
    for (int r = 0; r < 8; ++r) {
      x0[r] = sfr[0][r] * scale + pen0;
      x1[r] = sfr[1][r] * scale + pen1;
      float lm = fmaxf(x0[r], x1[r]);
      lm = fmaxf(lm, __shfl_xor(lm, 1, 32));
      lm = fmaxf(lm, __shfl_xor(lm, 2, 32));
      lm = fmaxf(lm, __shfl_xor(lm, 4, 32));
      lm = fmaxf(lm, __shfl_xor(lm, 8, 32));
      float nm = fmaxf(mrow[r], lm);
      corr[r] = __expf(mrow[r] - nm);
      mrow[r] = nm;
      x0[r] = __expf(x0[r] - nm);
      x1[r] = __expf(x1[r] - nm);
      float s = x0[r] + x1[r];
      s += __shfl_xor(s, 1, 32);
      s += __shfl_xor(s, 2, 32);
      s += __shfl_xor(s, 4, 32);
      s += __shfl_xor(s, 8, 32);
      lrow[r] = lrow[r] * corr[r] + s;
    }
#pragma unroll
    for (int t = 0; t < 4; ++t)
#pragma unroll
      for (int r = 0; r < 8; ++r) o[t][r] *= corr[r];

    // ---- C-layout -> A-layout for P, via per-wave LDS (in-order DS) ----
#pragma unroll
    for (int r = 0; r < 8; ++r) {
      int row = r + ((lane >> 4) << 3);
      plds[w][row][lane & 15]        = f2bf(x0[r]);
      plds[w][row][16 + (lane & 15)] = f2bf(x1[r]);
    }
    v16bf pa;
    {
      int m  = lane & 15;
      int kh = (lane >> 4) << 3;
      BF16x16 u;
      u.q[0] = *(const uint4*)(&plds[w][m][kh]);
      u.q[1] = *(const uint4*)(&plds[w][m][16 + kh]);
      pa = u.v;
    }

    // ---- O += P x V (Vt rows give contiguous B-frags) ----
#pragma unroll
    for (int t = 0; t < 4; ++t) {
      int dh = t * 16 + (lane & 15);
      const __bf16* vrow = vtb + (long)dh * cS + j + ((lane >> 4) << 3);
      BF16x16 u;
      u.q[0] = *(const uint4*)(vrow);
      u.q[1] = *(const uint4*)(vrow + 16);
      o[t] = wmma_bf16(pa, u.v, o[t]);
    }
  }

  // torch view quirk: flat (b*H+h) reinterpreted as (h2*B + b2)
  int f  = b * cH + h;
  int h2 = f / cB;
  int b2 = f % cB;
#pragma unroll
  for (int r = 0; r < 8; ++r) {
    int srow = q0 + r + ((lane >> 4) << 3);
    float inv = 1.0f / lrow[r];
    __bf16* orow = av + ((long)(b2 * cS + srow)) * cHD + h2 * cDH;
#pragma unroll
    for (int t = 0; t < 4; ++t)
      orow[t * 16 + (lane & 15)] = f2bf(o[t][r] * inv);
  }
}

// ---------------------------------------------------------------------------
// Residual + LayerNorm (+mask). Optionally writes f32 and bf16 (into padded
// conv-activation buffer at row 1+s of each batch slab).
// ---------------------------------------------------------------------------
__global__ __launch_bounds__(256)
void ln_residual(const float* __restrict__ x, const float* __restrict__ rz,
                 const float* __restrict__ g, const float* __restrict__ be,
                 const unsigned char* __restrict__ mk,
                 float* __restrict__ outf, __bf16* __restrict__ outb, int Dn)
{
  int row = blockIdx.x;
  int b = row / cS, s = row % cS;
  const float* xr = x  + (long)row * Dn;
  const float* rr = rz + (long)row * Dn;
  float sum = 0.f, sq = 0.f;
  for (int i = threadIdx.x; i < Dn; i += 256) {
    float y = xr[i] + rr[i];
    sum += y; sq += y * y;
  }
#pragma unroll
  for (int off = 16; off >= 1; off >>= 1) {
    sum += __shfl_xor(sum, off, 32);
    sq  += __shfl_xor(sq,  off, 32);
  }
  __shared__ float ssum[8], ssq[8];
  int w = threadIdx.x >> 5, lane = threadIdx.x & 31;
  if (lane == 0) { ssum[w] = sum; ssq[w] = sq; }
  __syncthreads();
  if (w == 0) {
    float a = lane < 8 ? ssum[lane] : 0.f;
    float q = lane < 8 ? ssq[lane]  : 0.f;
#pragma unroll
    for (int off = 4; off >= 1; off >>= 1) {
      a += __shfl_xor(a, off, 32);
      q += __shfl_xor(q, off, 32);
    }
    if (lane == 0) { ssum[0] = a; ssq[0] = q; }
  }
  __syncthreads();
  float mean = ssum[0] / Dn;
  float var  = ssq[0] / Dn - mean * mean;
  float rstd = rsqrtf(var + 1e-5f);
  float mval = mk[row] ? 1.f : 0.f;
  for (int i = threadIdx.x; i < Dn; i += 256) {
    float y = xr[i] + rr[i];
    float oy = ((y - mean) * rstd * g[i] + be[i]) * mval;
    if (outf) outf[(long)row * Dn + i] = oy;
    if (outb) outb[((long)b * (cS + 2) + 1 + s) * Dn + i] = f2bf(oy);
  }
}

// ---------------------------------------------------------------------------
// Small prep kernels
// ---------------------------------------------------------------------------
__global__ void cast_bf16(const float* __restrict__ in, __bf16* __restrict__ out, long n) {
  long i = (long)blockIdx.x * 256 + threadIdx.x;
  if (i < n) out[i] = f2bf(in[i]);
}
// W [rows,cols] f32 -> Wt [cols,rows] bf16
__global__ void transpose_cast(const float* __restrict__ in, __bf16* __restrict__ out,
                               int rows, int cols) {
  long i = (long)blockIdx.x * 256 + threadIdx.x;
  long n = (long)rows * cols;
  if (i >= n) return;
  int r = (int)(i / cols), c = (int)(i % cols);
  out[(long)c * rows + r] = f2bf(in[i]);
}
// conv w [N][C][K] f32 -> [N][K*C] bf16 with out[n][k*C+c] = w[n][c][k]
__global__ void repack_conv_w(const float* __restrict__ in, __bf16* __restrict__ out,
                              int N, int C, int K) {
  long i = (long)blockIdx.x * 256 + threadIdx.x;
  long n = (long)N * C * K;
  if (i >= n) return;
  int nn = (int)(i / ((long)C * K));
  int rem = (int)(i % ((long)C * K));
  int c = rem / K, k = rem % K;
  out[(long)nn * C * K + (long)k * C + c] = f2bf(in[i]);
}
// v [B*S, H*DH] bf16 -> vt [B,H,DH,S] bf16
__global__ void transpose_v(const __bf16* __restrict__ v, __bf16* __restrict__ vt) {
  long i = (long)blockIdx.x * 256 + threadIdx.x;
  long n = (long)cB * cH * cDH * cS;
  if (i >= n) return;
  int s = (int)(i % cS); long t = i / cS;
  int d = (int)(t % cDH); t /= cDH;
  int h = (int)(t % cH);  int b = (int)(t / cH);
  vt[i] = v[((long)(b * cS + s)) * cHD + h * cDH + d];
}
// zero the pad rows (row 0 and row S+1 of each batch slab)
__global__ void zero_pad_rows(__bf16* __restrict__ buf, long batchStride, long ld) {
  long idx = (long)blockIdx.x * 256 + threadIdx.x;
  if (idx >= ld) return;
  long base = blockIdx.z * batchStride + (blockIdx.y ? (long)(cS + 1) * ld : 0);
  buf[base + idx] = bf_zero();
}

// ---------------------------------------------------------------------------
extern "C" void kernel_launch(void* const* d_in, const int* in_sizes, int n_in,
                              void* d_out, int out_size, void* d_ws, size_t ws_size,
                              hipStream_t stream) {
  (void)in_sizes; (void)n_in; (void)out_size; (void)ws_size;
  const float*         dec  = (const float*)d_in[0];
  const unsigned char* mk   = (const unsigned char*)d_in[1];
  const float* Wq = (const float*)d_in[2];  const float* bq = (const float*)d_in[3];
  const float* Wk = (const float*)d_in[4];  const float* bk = (const float*)d_in[5];
  const float* Wv = (const float*)d_in[6];  const float* bv = (const float*)d_in[7];
  const float* Wo = (const float*)d_in[8];
  const float* ln1g = (const float*)d_in[9];  const float* ln1b = (const float*)d_in[10];
  const float* c1w  = (const float*)d_in[11]; const float* c1b  = (const float*)d_in[12];
  const float* c2w  = (const float*)d_in[13]; const float* c2b  = (const float*)d_in[14];
  const float* ln2g = (const float*)d_in[15]; const float* ln2b = (const float*)d_in[16];
  float* out = (float*)d_out;

  char* ws = (char*)d_ws; size_t off = 0;
  auto alloc = [&](size_t bytes) -> void* {
    void* p = ws + off; off = (off + bytes + 255) & ~(size_t)255; return p;
  };
  __bf16* xb   = (__bf16*)alloc((size_t)cBS * cD * 2);
  __bf16* wqT  = (__bf16*)alloc((size_t)cD * cHD * 2);
  __bf16* wkT  = (__bf16*)alloc((size_t)cD * cHD * 2);
  __bf16* wvT  = (__bf16*)alloc((size_t)cD * cHD * 2);
  __bf16* woT  = (__bf16*)alloc((size_t)cHD * cD * 2);
  __bf16* qb   = (__bf16*)alloc((size_t)cBS * cHD * 2);
  __bf16* kbf  = (__bf16*)alloc((size_t)cBS * cHD * 2);
  __bf16* vb   = (__bf16*)alloc((size_t)cBS * cHD * 2);
  __bf16* vt   = (__bf16*)alloc((size_t)cB * cH * cDH * cS * 2);
  __bf16* av   = (__bf16*)alloc((size_t)cBS * cHD * 2);
  float*  attn = (float*) alloc((size_t)cBS * cD * 4);
  float*  out1 = (float*) alloc((size_t)cBS * cD * 4);
  __bf16* ypad = (__bf16*)alloc((size_t)cB * (cS + 2) * cD * 2);
  __bf16* w1t  = (__bf16*)alloc((size_t)cDI * 3 * cD * 2);
  __bf16* h1p  = (__bf16*)alloc((size_t)cB * (cS + 2) * cDI * 2);
  __bf16* w2t  = (__bf16*)alloc((size_t)cD * 3 * cDI * 2);
  float*  h2   = (float*) alloc((size_t)cBS * cD * 4);

  dim3 blk(256);
  auto g1 = [](long n) { return dim3((unsigned)((n + 255) / 256)); };

  // ---- prep: casts / transposes / conv repack ----
  cast_bf16<<<g1((long)cBS * cD), blk, 0, stream>>>(dec, xb, (long)cBS * cD);
  transpose_cast<<<g1((long)cD * cHD), blk, 0, stream>>>(Wq, wqT, cD, cHD);
  transpose_cast<<<g1((long)cD * cHD), blk, 0, stream>>>(Wk, wkT, cD, cHD);
  transpose_cast<<<g1((long)cD * cHD), blk, 0, stream>>>(Wv, wvT, cD, cHD);
  transpose_cast<<<g1((long)cHD * cD), blk, 0, stream>>>(Wo, woT, cHD, cD);
  repack_conv_w<<<g1((long)cDI * cD * 3), blk, 0, stream>>>(c1w, w1t, cDI, cD, 3);
  repack_conv_w<<<g1((long)cD * cDI * 3), blk, 0, stream>>>(c2w, w2t, cD, cDI, 3);

  // ---- QKV projections: [4096,1024] x [1024,1024] -> bf16 ----
  dim3 gqkv(cHD / 256, cBS / 64, 1);
  gemm_bf16_wmma<<<gqkv, blk, 0, stream>>>(xb, cD, 0, wqT, cD, bq,
                                           nullptr, 0, 0, qb, cHD, 0,
                                           cBS, cHD, cD, 0);
  gemm_bf16_wmma<<<gqkv, blk, 0, stream>>>(xb, cD, 0, wkT, cD, bk,
                                           nullptr, 0, 0, kbf, cHD, 0,
                                           cBS, cHD, cD, 0);
  gemm_bf16_wmma<<<gqkv, blk, 0, stream>>>(xb, cD, 0, wvT, cD, bv,
                                           nullptr, 0, 0, vb, cHD, 0,
                                           cBS, cHD, cD, 0);
  transpose_v<<<g1((long)cB * cH * cDH * cS), blk, 0, stream>>>(vb, vt);

  // ---- flash attention ----
  attn_flash<<<dim3(cS / 128, cH, cB), blk, 0, stream>>>(qb, kbf, vt, mk, av, 0.125f);

  // ---- output projection (no bias) -> f32 attn ----
  gemm_bf16_wmma<<<dim3(cD / 256, cBS / 64, 1), blk, 0, stream>>>(
      av, cHD, 0, woT, cHD, nullptr,
      attn, cD, 0, nullptr, 0, 0, cBS, cD, cHD, 0);

  // ---- residual + LN1 (+mask) -> out1 f32 and padded bf16 ----
  zero_pad_rows<<<dim3((cD + 255) / 256, 2, cB), blk, 0, stream>>>(
      ypad, (long)(cS + 2) * cD, cD);
  ln_residual<<<cBS, blk, 0, stream>>>(dec, attn, ln1g, ln1b, mk, out1, ypad, cD);

  // ---- conv1 as implicit-im2col GEMM: K = 3*D, overlapping padded rows ----
  zero_pad_rows<<<dim3((cDI + 255) / 256, 2, cB), blk, 0, stream>>>(
      h1p, (long)(cS + 2) * cDI, cDI);
  gemm_bf16_wmma<<<dim3(cDI / 256, cS / 64, cB), blk, 0, stream>>>(
      ypad, cD, (long)(cS + 2) * cD, w1t, 3 * cD, c1b,
      nullptr, 0, 0, h1p + cDI, cDI, (long)(cS + 2) * cDI,
      cS, cDI, 3 * cD, /*relu=*/1);

  // ---- conv2: K = 3*DI ----
  gemm_bf16_wmma<<<dim3(cD / 256, cS / 64, cB), blk, 0, stream>>>(
      h1p, cDI, (long)(cS + 2) * cDI, w2t, 3 * cDI, c2b,
      h2, cD, (long)cS * cD, nullptr, 0, 0,
      cS, cD, 3 * cDI, 0);

  // ---- residual + LN2 (+mask) -> final output ----
  ln_residual<<<cBS, blk, 0, stream>>>(out1, h2, ln2g, ln2b, mk, out, nullptr, cD);
}
